// LocalGlobalAttention_11003706212960
// MI455X (gfx1250) — matrix operand order repaired and checked by reading, not compiled, not run on hardware
//
#include <hip/hip_runtime.h>

typedef __bf16 bf16_t;
typedef bf16_t v16bf __attribute__((ext_vector_type(16)));
typedef bf16_t v8bf  __attribute__((ext_vector_type(8)));
typedef float  v8f   __attribute__((ext_vector_type(8)));

#define T_SEQ   4096
#define D_HEAD  64
#define WIN     512
#define GSTRIDE 64
#define QTILE   64
#define KTILE   64
#define NWAVES  4
#define KSTR    72   // padded LDS strides (bf16): 144B rows -> 16 readers tile banks
#define VSTR    72
#define PSTR    72

// DPP8 selector for xor-permute within 8-lane groups
constexpr unsigned dpp8_xor(int m) {
  unsigned s = 0;
  for (int i = 0; i < 8; ++i) s |= (unsigned)(i ^ m) << (3 * i);
  return s;
}
// raw max (llvm.maxnum inserts canonicalize; our data has no sNaNs)
static __device__ __forceinline__ float vmax(float a, float b) {
  float r;
  asm("v_max_num_f32 %0, %1, %2" : "=v"(r) : "v"(a), "v"(b));
  return r;
}
// reduce max over each 16-lane half: xor1/2/4 via DPP8, xor8 via bpermute
static __device__ __forceinline__ float halfmax_red(float r) {
  float t;
  t = __builtin_bit_cast(float, __builtin_amdgcn_mov_dpp8(__builtin_bit_cast(int, r), dpp8_xor(1)));
  r = vmax(r, t);
  t = __builtin_bit_cast(float, __builtin_amdgcn_mov_dpp8(__builtin_bit_cast(int, r), dpp8_xor(2)));
  r = vmax(r, t);
  t = __builtin_bit_cast(float, __builtin_amdgcn_mov_dpp8(__builtin_bit_cast(int, r), dpp8_xor(4)));
  r = vmax(r, t);
  r = vmax(r, __shfl_xor(r, 8, 32));
  return r;
}
// 16 consecutive bf16 (32 B) from LDS -> one B/A fragment (two ds_load_b128)
static __device__ __forceinline__ v16bf ld32B(const bf16_t* p) {
  v8bf lo = *(const v8bf*)p;
  v8bf hi = *(const v8bf*)(p + 8);
  v16bf r;
#pragma unroll
  for (int i = 0; i < 8; ++i) { r[i] = lo[i]; r[8 + i] = hi[i]; }
  return r;
}
// two disjoint 16B runs (A-fragment halves K=kbA.. and K=16+kbA..)
static __device__ __forceinline__ v16bf ld2x16B(const bf16_t* p0, const bf16_t* p1) {
  v8bf lo = *(const v8bf*)p0;
  v8bf hi = *(const v8bf*)p1;
  v16bf r;
#pragma unroll
  for (int i = 0; i < 8; ++i) { r[i] = lo[i]; r[8 + i] = hi[i]; }
  return r;
}

__global__ __launch_bounds__(128, 1)
void lga_fwd_kernel(const float* __restrict__ Qg, const float* __restrict__ Kg,
                    const float* __restrict__ Vg, float* __restrict__ Og) {
  __shared__ bf16_t Ksh[2][KTILE * KSTR];      // ping-pong K tiles (18.4 KB)
  __shared__ bf16_t VTsh[2][64 * VSTR];        // ping-pong V^T tiles (18.4 KB)
  __shared__ bf16_t Psh[NWAVES][16 * PSTR];    // per-wave P staging (9.2 KB)

  const int tid   = threadIdx.x;
  const int lane  = tid & 31;
  const int wave  = tid >> 5;
  const int q0    = blockIdx.x * QTILE;
  const int head  = blockIdx.y;
  const int qrow0 = q0 + wave * 16;

  const int col = lane & 15;   // N (C/B layouts) / M row (A layout)
  const int hlf = lane >> 4;
  const int kbA = hlf * 8;     // A-frag K base within 16-K group
  const int kbB = hlf * 16;    // unused: B frags use kbA, same k order as A

  // staging roles: 128 threads x 2 passes cover 64 rows x 4 d-segments
  const int srow = tid >> 2;
  const int sseg = (tid & 3) << 4;

  const size_t hb = (size_t)head * T_SEQ * D_HEAD;
  const float* Qh = Qg + hb;
  const float* Kh = Kg + hb;
  const float* Vh = Vg + hb;
  float*       Oh = Og + hb;

  const float sc  = 0.125f * 1.44269504088896340736f; // 1/sqrt(64)*log2(e), folded into Q
  const float NEG = -__builtin_inff();

  // ---- Q strip (16x64), pre-scaled, resident as two bf16 A-fragments ----
  v16bf qa0, qa1;
  {
    const float* qp = Qh + (size_t)(qrow0 + col) * D_HEAD;
#pragma unroll
    for (int i = 0; i < 8; ++i) {
      qa0[i]     = (bf16_t)(qp[kbA + i] * sc);
      qa0[8 + i] = (bf16_t)(qp[16 + kbA + i] * sc);
      qa1[i]     = (bf16_t)(qp[32 + kbA + i] * sc);
      qa1[8 + i] = (bf16_t)(qp[48 + kbA + i] * sc);
    }
  }

  // ones-column B fragment: B[k][0] = 1, else 0  ->  accL column 0 accumulates row-sums
  v16bf vOne;
#pragma unroll
  for (int i = 0; i < 16; ++i) vOne[i] = (col == 0) ? (bf16_t)1.0f : (bf16_t)0.0f;

  float mrow[8];
  v8f acc0 = {}, acc1 = {}, acc2 = {}, acc3 = {}, accL = {};
#pragma unroll
  for (int r = 0; r < 8; ++r) mrow[r] = -1e30f;

  const int kstart = (q0 > WIN) ? (q0 - WIN) : 0;   // multiple of 64
  const int nglob  = kstart / GSTRIDE;              // <= 55 -> at most one global tile

  auto tile = [&](int kbase, bool gphase, int gcount, int buf) {
    // ---- cooperative staging: 64 rows, two per thread ----
#pragma unroll
    for (int sh = 0; sh < 2; ++sh) {
      const int row = srow + sh * 32;
      int j = kbase + row;
      int key = gphase ? ((j < gcount) ? j * GSTRIDE : 0) : j;
      const float4* kp4 = (const float4*)(Kh + (size_t)key * D_HEAD + sseg);
      const float4* vp4 = (const float4*)(Vh + (size_t)key * D_HEAD + sseg);
      float4 k0 = kp4[0], k1 = kp4[1], k2 = kp4[2], k3 = kp4[3];
      float4 w0 = vp4[0], w1 = vp4[1], w2 = vp4[2], w3 = vp4[3];
      {
        int jn = j + KTILE;   // next-tile prefetch (first-pass HBM latency)
        int keyn = gphase ? ((jn < gcount) ? jn * GSTRIDE : 0)
                          : ((jn < T_SEQ) ? jn : 0);
        __builtin_prefetch(Kh + (size_t)keyn * D_HEAD + sseg, 0, 3);
        __builtin_prefetch(Vh + (size_t)keyn * D_HEAD + sseg, 0, 3);
      }
      v8bf lo, hi;
      lo[0] = (bf16_t)k0.x; lo[1] = (bf16_t)k0.y; lo[2] = (bf16_t)k0.z; lo[3] = (bf16_t)k0.w;
      lo[4] = (bf16_t)k1.x; lo[5] = (bf16_t)k1.y; lo[6] = (bf16_t)k1.z; lo[7] = (bf16_t)k1.w;
      hi[0] = (bf16_t)k2.x; hi[1] = (bf16_t)k2.y; hi[2] = (bf16_t)k2.z; hi[3] = (bf16_t)k2.w;
      hi[4] = (bf16_t)k3.x; hi[5] = (bf16_t)k3.y; hi[6] = (bf16_t)k3.z; hi[7] = (bf16_t)k3.w;
      *(v8bf*)&Ksh[buf][row * KSTR + sseg]     = lo;
      *(v8bf*)&Ksh[buf][row * KSTR + sseg + 8] = hi;
      float vv[16] = {w0.x, w0.y, w0.z, w0.w, w1.x, w1.y, w1.z, w1.w,
                      w2.x, w2.y, w2.z, w2.w, w3.x, w3.y, w3.z, w3.w};
#pragma unroll
      for (int d = 0; d < 16; ++d)
        VTsh[buf][(sseg + d) * VSTR + row] = (bf16_t)vv[d];
    }
    __syncthreads();   // sole barrier: ping-pong makes read-vs-next-write safe

    // ---- QK^T: 4 key groups of 16 ----
    v8f s[4];
#pragma unroll
    for (int g = 0; g < 4; ++g) {
      const bf16_t* kr = &Ksh[buf][(g * 16 + col) * KSTR];
      v16bf kfa = ld2x16B(kr + kbA, kr + 16 + kbA);
      v16bf kfb = ld2x16B(kr + 32 + kbA, kr + 48 + kbA);
      v8f t = {};
      t = __builtin_amdgcn_wmma_f32_16x16x32_bf16(false, qa0, false, kfa, (short)0, t, false, false);
      t = __builtin_amdgcn_wmma_f32_16x16x32_bf16(false, qa1, false, kfb, (short)0, t, false, false);
      s[g] = t;
    }

    // ---- mask (scores already in scaled-log2 domain) ----
    if (gphase) {
#pragma unroll
      for (int g = 0; g < 4; ++g) {
        const bool ok = (kbase + g * 16 + col) < gcount;
#pragma unroll
        for (int r = 0; r < 8; ++r) s[g][r] = ok ? s[g][r] : NEG;
      }
    } else {
#pragma unroll
      for (int g = 0; g < 4; ++g) {
        const int key = kbase + g * 16 + col;
        const bool glb = (key & (GSTRIDE - 1)) == 0;
#pragma unroll
        for (int r = 0; r < 8; ++r) {
          const int q = qrow0 + r + 8 * hlf;
          bool m = (key <= q) && ((q - key <= WIN) || glb);
          s[g][r] = m ? s[g][r] : NEG;
        }
      }
    }

    // ---- online softmax over 64 keys at once ----
    v8f av;
#pragma unroll
    for (int r = 0; r < 8; ++r) {
      float t = vmax(vmax(s[0][r], s[1][r]), vmax(s[2][r], s[3][r]));
      t = halfmax_red(t);
      float mn = vmax(mrow[r], t);                     // finite: mrow >= -1e30
      av[r] = __builtin_amdgcn_exp2f(mrow[r] - mn);
      s[0][r] = __builtin_amdgcn_exp2f(s[0][r] - mn);
      s[1][r] = __builtin_amdgcn_exp2f(s[1][r] - mn);
      s[2][r] = __builtin_amdgcn_exp2f(s[2][r] - mn);
      s[3][r] = __builtin_amdgcn_exp2f(s[3][r] - mn);
      mrow[r] = mn;
    }
    acc0 *= av; acc1 *= av; acc2 *= av; acc3 *= av; accL *= av;

    // ---- P: C layout -> A layout via per-wave LDS (in-order DS within wave) ----
    bf16_t* myP = &Psh[wave][0];
#pragma unroll
    for (int g = 0; g < 4; ++g)
#pragma unroll
      for (int r = 0; r < 8; ++r)
        myP[(r + 8 * hlf) * PSTR + g * 16 + col] = (bf16_t)s[g][r];
    __builtin_amdgcn_wave_barrier();   // pin compiler ordering; HW DS is in-order
    const bf16_t* pp = &Psh[wave][col * PSTR];
    v16bf pa0 = ld2x16B(pp + kbA,      pp + 16 + kbA);
    v16bf pa1 = ld2x16B(pp + 32 + kbA, pp + 48 + kbA);

    // ---- O += P * V, two 32-key halves; row-sums into accL column 0 ----
#pragma unroll
    for (int sh = 0; sh < 2; ++sh) {
      const v16bf pa = sh ? pa1 : pa0;
      const int ko = sh * 32 + kbA;
      v16bf vb0 = ld2x16B(&VTsh[buf][(col)      * VSTR + ko], &VTsh[buf][(col)      * VSTR + ko + 16]);
      v16bf vb1 = ld2x16B(&VTsh[buf][(16 + col) * VSTR + ko], &VTsh[buf][(16 + col) * VSTR + ko + 16]);
      v16bf vb2 = ld2x16B(&VTsh[buf][(32 + col) * VSTR + ko], &VTsh[buf][(32 + col) * VSTR + ko + 16]);
      v16bf vb3 = ld2x16B(&VTsh[buf][(48 + col) * VSTR + ko], &VTsh[buf][(48 + col) * VSTR + ko + 16]);
      acc0 = __builtin_amdgcn_wmma_f32_16x16x32_bf16(false, pa, false, vb0, (short)0, acc0, false, false);
      acc1 = __builtin_amdgcn_wmma_f32_16x16x32_bf16(false, pa, false, vb1, (short)0, acc1, false, false);
      acc2 = __builtin_amdgcn_wmma_f32_16x16x32_bf16(false, pa, false, vb2, (short)0, acc2, false, false);
      acc3 = __builtin_amdgcn_wmma_f32_16x16x32_bf16(false, pa, false, vb3, (short)0, acc3, false, false);
      accL = __builtin_amdgcn_wmma_f32_16x16x32_bf16(false, pa, false, vOne, (short)0, accL, false, false);
    }
  };

  // Phase 1: pure-global strided keys (k = 64*j, j < nglob) -> at most one tile
  // Phase 2: local window + diagonal (in-window global keys handled by OR mask)
  int par = 0;
  for (int jb = 0; jb < nglob; jb += KTILE, ++par) tile(jb, true, nglob, par & 1);
  for (int kb = kstart; kb < q0 + QTILE; kb += KTILE, ++par) tile(kb, false, 0, par & 1);

  // ---- epilogue: O = acc / l, l broadcast from column 0 of accL ----
  v8f invv;
#pragma unroll
  for (int r = 0; r < 8; ++r)
    invv[r] = 1.0f / __shfl(accL[r], 0, 16);   // lane 0 / 16 of each half
  acc0 *= invv; acc1 *= invv; acc2 *= invv; acc3 *= invv;
#pragma unroll
  for (int r = 0; r < 8; ++r) {
    const int q = qrow0 + r + 8 * hlf;
    float* op = Oh + (size_t)q * D_HEAD + col;
    op[0]  = acc0[r];
    op[16] = acc1[r];
    op[32] = acc2[r];
    op[48] = acc3[r];
  }
}

extern "C" void kernel_launch(void* const* d_in, const int* in_sizes, int n_in,
                              void* d_out, int out_size, void* d_ws, size_t ws_size,
                              hipStream_t stream) {
  (void)n_in; (void)out_size; (void)d_ws; (void)ws_size;
  const float* Q = (const float*)d_in[0];
  const float* K = (const float*)d_in[1];
  const float* V = (const float*)d_in[2];
  float* O = (float*)d_out;
  const int H = in_sizes[0] / (T_SEQ * D_HEAD);   // B=1, H=8 here
  dim3 grid(T_SEQ / QTILE, H, 1);
  dim3 block(128, 1, 1);
  lga_fwd_kernel<<<grid, block, 0, stream>>>(Q, K, V, O);
}
